// MessageDecoder_51685636440348
// MI455X (gfx1250) — compile-verified, matrix operand order checked by reading, not run on hardware
//
#include <hip/hip_runtime.h>
#include <hip/hip_bf16.h>
#include <math.h>

typedef _Float16 f16;
typedef __attribute__((ext_vector_type(16))) _Float16 v16h;
typedef __attribute__((ext_vector_type(8)))  _Float16 v8h;
typedef __attribute__((ext_vector_type(8)))  float    v8f;

#define BATCH   32768
#define HDIM    512
#define G4H     2048     // 4*H
#define MAXLEN  20
#define NACT    65
#define NPAD    80       // 65 padded to 5 WMMA col tiles
#define VOCAB   67
#define PADTOK  65
#define BOSTOK  66
#define EOSTOK  0

__device__ __forceinline__ float sig_(float x) { return 1.0f / (1.0f + expf(-x)); }

__device__ __forceinline__ v8f zero8() {
    v8f z;
#pragma unroll
    for (int i = 0; i < 8; ++i) z[i] = 0.0f;
    return z;
}

// Load one lane's slice of a 16x32 f16 WMMA operand tile from global memory.
// ISA layout (CDNA5 ISA, 16-bit A 16x32): lane L (row/col = L&15, g = L>>4)
// holds halfs t=0..7 -> K = 8g + t, t=8..15 -> K = 16 + 8g + (t-8):
// i.e. two contiguous 16-byte chunks at k0+8g and k0+16+8g.
__device__ __forceinline__ v16h load_tile16(const f16* __restrict__ rowk, int g) {
    const v8h lo = *reinterpret_cast<const v8h*>(rowk + 8 * g);
    const v8h hi = *reinterpret_cast<const v8h*>(rowk + 16 + 8 * g);
    v16h a;
#pragma unroll
    for (int i = 0; i < 8; ++i) { a[i] = lo[i]; a[i + 8] = hi[i]; }
    return a;
}

// Same tile fetch but from an XOR-swizzled LDS row (16B-chunk index ^ (row&15)).
__device__ __forceinline__ v16h load_tile16_lds(const f16* base_row, int kk, int g, int x) {
    const int c0 = (kk >> 3) + g;        // 16B chunk indices within the row
    const int c1 = c0 + 2;
    const v8h lo = *reinterpret_cast<const v8h*>(base_row + ((c0 ^ x) << 3));
    const v8h hi = *reinterpret_cast<const v8h*>(base_row + ((c1 ^ x) << 3));
    v16h a;
#pragma unroll
    for (int i = 0; i < 8; ++i) { a[i] = lo[i]; a[i + 8] = hi[i]; }
    return a;
}

// ---------------- prep kernels ----------------

__global__ void __launch_bounds__(256) cvt_kernel(const float* __restrict__ s,
                                                  f16* __restrict__ d, int n) {
    int i = blockIdx.x * 256 + threadIdx.x;
    if (i < n) d[i] = (f16)s[i];
}

__global__ void __launch_bounds__(256) cvt_wa_kernel(const float* __restrict__ Wa,
                                                     f16* __restrict__ d) {
    int i = blockIdx.x * 256 + threadIdx.x;
    if (i >= NPAD * HDIM) return;
    int n = i / HDIM, k = i % HDIM;
    d[i] = (f16)((n < NACT) ? Wa[n * HDIM + k] : 0.0f);
}

// xtable[v][n] = b_ih[n] + b_hh[n] + sum_k emb[v][k] * W_ih[n][k]
__global__ void __launch_bounds__(256) xtab_kernel(const float* __restrict__ emb,
                                                   const float* __restrict__ Wih,
                                                   const float* __restrict__ bih,
                                                   const float* __restrict__ bhh,
                                                   float* __restrict__ xt) {
    int i = blockIdx.x * 256 + threadIdx.x;
    if (i >= VOCAB * G4H) return;
    int v = i / G4H, n = i % G4H;
    const float* e = emb + (size_t)v * HDIM;
    const float* w = Wih + (size_t)n * HDIM;
    float acc = bih[n] + bhh[n];
#pragma unroll 4
    for (int k = 0; k < HDIM; ++k) acc += e[k] * w[k];
    xt[i] = acc;
}

__global__ void __launch_bounds__(256) initstate_kernel(int* __restrict__ last,
                                                        int* __restrict__ stopped) {
    int i = blockIdx.x * 256 + threadIdx.x;
    if (i < BATCH) { last[i] = BOSTOK; stopped[i] = 0; }
}

// ---------------- h0 / c0 projection (WMMA) ----------------
// sel=0: h0 = enc @ Wc.T + bc  (f16 out)   sel=1: c0 = enc @ Wh.T + bh (f32 out)
__global__ void __launch_bounds__(256) proj_kernel(const f16* __restrict__ enc,
                                                   const f16* __restrict__ Wc16,
                                                   const f16* __restrict__ Wh16,
                                                   const float* __restrict__ bc,
                                                   const float* __restrict__ bh,
                                                   f16* __restrict__ h0,
                                                   float* __restrict__ c0) {
    const int wave = (blockIdx.x * 256 + threadIdx.x) >> 5;
    const int lane = threadIdx.x & 31;
    const int sel = wave >> 14;          // 16384 waves per matrix (M-block of 4)
    const int inner = wave & 16383;
    const int m0 = (inner >> 5) * 64;    // 512 m-groups of 64 rows
    const int j0 = (inner & 31) * 16;
    const int r15 = lane & 15, g = lane >> 4;

    const f16* W = sel ? Wh16 : Wc16;
    const float* bias = sel ? bh : bc;
    const f16* brow = W + (size_t)(j0 + r15) * HDIM;

    v8f acc[4];
#pragma unroll
    for (int m = 0; m < 4; ++m) acc[m] = zero8();

    for (int kk = 0; kk < HDIM; kk += 32) {
        v16h b = load_tile16(brow + kk, g);
#pragma unroll
        for (int m = 0; m < 4; ++m) {
            v16h a = load_tile16(enc + (size_t)(m0 + m * 16 + r15) * HDIM + kk, g);
            acc[m] = __builtin_amdgcn_wmma_f32_16x16x32_f16(false, a, false, b,
                                                            (short)0, acc[m], false, false);
        }
    }
    const int col = j0 + r15;
    const float bv = bias[col];
#pragma unroll
    for (int m = 0; m < 4; ++m) {
#pragma unroll
        for (int r = 0; r < 8; ++r) {
            const int bi = m0 + m * 16 + r + 8 * g;
            const size_t off = (size_t)bi * HDIM + col;
            const float v = acc[m][r] + bv;
            if (sel == 0) h0[off] = (f16)v;
            else          c0[off] = v;
        }
    }
}

// ---------------- LSTM step (WMMA GEMM + fused elementwise) ----------------
// Block: 64 batch rows staged once into a 64KB XOR-swizzled LDS panel, shared
// by all 8 waves (one 16-col j-tile each). Each wave: 64 rows x 16 cols x 4
// gate quadrants -> 16 accumulators; B (W_hh) tiles reused across 4 A tiles.
__global__ void __launch_bounds__(256) lstm_step_kernel(const f16* __restrict__ hin,
                                                        f16* __restrict__ hout,
                                                        float* __restrict__ c,
                                                        const f16* __restrict__ Whh,
                                                        const float* __restrict__ xtab,
                                                        const int* __restrict__ last) {
    __shared__ f16 Atile[64][HDIM];      // exactly 64KB, XOR-swizzled rows
    const int wv = threadIdx.x >> 5;
    const int lane = threadIdx.x & 31;
    const int m0 = (blockIdx.x >> 2) * 64;              // 512 m-groups
    const int j0 = ((blockIdx.x & 3) * 8 + wv) * 16;    // 32 j tiles, 8 per block
    const int r15 = lane & 15, g = lane >> 4;

    // cooperative global -> LDS stage of the 64-row A panel (16B chunks)
#pragma unroll
    for (int i = 0; i < 16; ++i) {
        const int cidx = i * 256 + threadIdx.x;         // 4096 chunks of 16B
        const int row = cidx >> 6;                      // 64 chunks per row
        const int cc = cidx & 63;
        const v8h d = *reinterpret_cast<const v8h*>(
            hin + (size_t)(m0 + row) * HDIM + cc * 8);
        *reinterpret_cast<v8h*>(&Atile[row][(cc ^ (row & 15)) << 3]) = d;
    }
    __syncthreads();

    v8f acc[4][4];                        // [m][gate]
#pragma unroll
    for (int m = 0; m < 4; ++m)
#pragma unroll
        for (int q = 0; q < 4; ++q) acc[m][q] = zero8();

    for (int kk = 0; kk < HDIM; kk += 32) {
        v16h b[4];
#pragma unroll
        for (int q = 0; q < 4; ++q)
            b[q] = load_tile16(Whh + (size_t)(q * HDIM + j0 + r15) * HDIM + kk, g);
#pragma unroll
        for (int m = 0; m < 4; ++m) {
            v16h a = load_tile16_lds(&Atile[m * 16 + r15][0], kk, g, r15);
#pragma unroll
            for (int q = 0; q < 4; ++q)
                acc[m][q] = __builtin_amdgcn_wmma_f32_16x16x32_f16(false, a, false, b[q],
                                                                   (short)0, acc[m][q],
                                                                   false, false);
        }
    }

    const int col = j0 + r15;
#pragma unroll
    for (int m = 0; m < 4; ++m) {
#pragma unroll
        for (int r = 0; r < 8; ++r) {
            const int bi = m0 + m * 16 + r + 8 * g;
            const int li = last[bi];
            const float* xr = xtab + (size_t)li * G4H;
            const float gi = acc[m][0][r] + xr[col];
            const float gf = acc[m][1][r] + xr[col + HDIM];
            const float gg = acc[m][2][r] + xr[col + 2 * HDIM];
            const float go = acc[m][3][r] + xr[col + 3 * HDIM];
            const size_t off = (size_t)bi * HDIM + col;
            const float cn = sig_(gf) * c[off] + sig_(gi) * tanhf(gg);
            c[off] = cn;
            hout[off] = (f16)(sig_(go) * tanhf(cn));
        }
    }
}

// ---------------- logits + log-softmax/argmax/entropy ----------------
// 128-thread blocks (4 waves); each wave: 32 batch rows x 80 cols (2 M-tiles,
// 5 col tiles). All 32 lanes then reduce one row each through LDS.
__global__ void __launch_bounds__(128) logits_kernel(const f16* __restrict__ h,
                                                     const f16* __restrict__ Wa16,
                                                     const float* __restrict__ ba,
                                                     int* __restrict__ last,
                                                     int* __restrict__ stopped,
                                                     float* __restrict__ msgb,
                                                     float* __restrict__ lpb,
                                                     float* __restrict__ entb,
                                                     int step) {
    __shared__ float lds[4][32][NPAD + 4];
    const int wv = threadIdx.x >> 5;
    const int lane = threadIdx.x & 31;
    const int wave = blockIdx.x * 4 + wv;
    const int m0 = wave * 32;
    const int r15 = lane & 15, g = lane >> 4;

    v8f acc[2][5];
#pragma unroll
    for (int m = 0; m < 2; ++m)
#pragma unroll
        for (int t = 0; t < 5; ++t) acc[m][t] = zero8();

    for (int kk = 0; kk < HDIM; kk += 32) {
        v16h b[5];
#pragma unroll
        for (int t = 0; t < 5; ++t)
            b[t] = load_tile16(Wa16 + (size_t)(t * 16 + r15) * HDIM + kk, g);
#pragma unroll
        for (int m = 0; m < 2; ++m) {
            v16h a = load_tile16(h + (size_t)(m0 + m * 16 + r15) * HDIM + kk, g);
#pragma unroll
            for (int t = 0; t < 5; ++t)
                acc[m][t] = __builtin_amdgcn_wmma_f32_16x16x32_f16(false, a, false, b[t],
                                                                   (short)0, acc[m][t],
                                                                   false, false);
        }
    }
#pragma unroll
    for (int m = 0; m < 2; ++m)
#pragma unroll
        for (int t = 0; t < 5; ++t)
#pragma unroll
            for (int r = 0; r < 8; ++r)
                lds[wv][m * 16 + r + 8 * g][t * 16 + r15] = acc[m][t][r];
    __syncthreads();

    {
        const int bi = m0 + lane;         // one row per lane (32 rows/wave)
        float* lrow = lds[wv][lane];
        float maxv = -3.0e38f;
        int bestj = 0;
        for (int j = 0; j < NACT; ++j) {
            float v = lrow[j] + ba[j];
            lrow[j] = v;
            if (v > maxv) { maxv = v; bestj = j; }   // first-max tie-break
        }
        float se = 0.0f, sl = 0.0f;
        for (int j = 0; j < NACT; ++j) {
            float p = expf(lrow[j] - maxv);
            se += p;
            sl += p * lrow[j];
        }
        const float logZ = maxv + logf(se);
        const float ent = logZ - sl / se;
        const float lp = lrow[bestj] - logZ;

        const int st = stopped[bi];
        const float alive = st ? 0.0f : 1.0f;
        int action = st ? PADTOK : bestj;
        const int nst = st | (action == EOSTOK);

        msgb[(size_t)step * BATCH + bi] = (float)action;
        lpb [(size_t)step * BATCH + bi] = lp * alive;
        entb[(size_t)step * BATCH + bi] = ent * alive;
        last[bi] = action;
        stopped[bi] = nst;
    }
}

// ---------------- assemble outputs ----------------
// d_out: entropy[B] | log_probs[B,20] | message[B,20] | message_len[B]
__global__ void __launch_bounds__(256) finalize_kernel(const float* __restrict__ msgb,
                                                       const float* __restrict__ lpb,
                                                       const float* __restrict__ entb,
                                                       float* __restrict__ out) {
    int b = blockIdx.x * 256 + threadIdx.x;
    if (b >= BATCH) return;
    float esum = 0.0f;
    int len = 0;
#pragma unroll
    for (int s = 0; s < MAXLEN; ++s) {
        const float m = msgb[(size_t)s * BATCH + b];
        if (m != (float)PADTOK) ++len;
        esum += entb[(size_t)s * BATCH + b];
        out[BATCH + (size_t)b * MAXLEN + s] = lpb[(size_t)s * BATCH + b];
        out[BATCH + (size_t)MAXLEN * BATCH + (size_t)b * MAXLEN + s] = m;
    }
    out[b] = esum / (float)len;
    out[BATCH + (size_t)2 * MAXLEN * BATCH + b] = (float)len;
}

extern "C" void kernel_launch(void* const* d_in, const int* in_sizes, int n_in,
                              void* d_out, int out_size, void* d_ws, size_t ws_size,
                              hipStream_t stream) {
    (void)in_sizes; (void)n_in; (void)out_size; (void)ws_size;
    const float* encoded = (const float*)d_in[0];
    const float* emb     = (const float*)d_in[1];
    const float* W_ih    = (const float*)d_in[2];
    const float* W_hh    = (const float*)d_in[3];
    const float* b_ih    = (const float*)d_in[4];
    const float* b_hh    = (const float*)d_in[5];
    const float* Wc      = (const float*)d_in[6];
    const float* bc      = (const float*)d_in[7];
    const float* Wh      = (const float*)d_in[8];
    const float* bh      = (const float*)d_in[9];
    const float* Wa      = (const float*)d_in[10];
    const float* ba      = (const float*)d_in[11];
    float* out = (float*)d_out;

    char* w = (char*)d_ws;
    f16*   hA      = (f16*)w;   w += (size_t)BATCH * HDIM * 2;   // 32 MB
    f16*   hB      = (f16*)w;   w += (size_t)BATCH * HDIM * 2;   // 32 MB (also enc_f16)
    float* cbuf    = (float*)w; w += (size_t)BATCH * HDIM * 4;   // 64 MB
    f16*   Whh16   = (f16*)w;   w += (size_t)G4H * HDIM * 2;
    f16*   Wc16    = (f16*)w;   w += (size_t)HDIM * HDIM * 2;
    f16*   Wh16    = (f16*)w;   w += (size_t)HDIM * HDIM * 2;
    f16*   Wa16    = (f16*)w;   w += (size_t)NPAD * HDIM * 2;
    float* xtab    = (float*)w; w += (size_t)VOCAB * G4H * 4;
    int*   last    = (int*)w;   w += (size_t)BATCH * 4;
    int*   stopped = (int*)w;   w += (size_t)BATCH * 4;
    float* msgb    = (float*)w; w += (size_t)MAXLEN * BATCH * 4;
    float* lpb     = (float*)w; w += (size_t)MAXLEN * BATCH * 4;
    float* entb    = (float*)w; w += (size_t)MAXLEN * BATCH * 4;
    f16*   enc16   = hB;  // consumed by proj before step 0 overwrites hB

    // prep
    cvt_kernel<<<(G4H * HDIM + 255) / 256, 256, 0, stream>>>(W_hh, Whh16, G4H * HDIM);
    cvt_kernel<<<(HDIM * HDIM + 255) / 256, 256, 0, stream>>>(Wc, Wc16, HDIM * HDIM);
    cvt_kernel<<<(HDIM * HDIM + 255) / 256, 256, 0, stream>>>(Wh, Wh16, HDIM * HDIM);
    cvt_wa_kernel<<<(NPAD * HDIM + 255) / 256, 256, 0, stream>>>(Wa, Wa16);
    cvt_kernel<<<(BATCH * HDIM + 255) / 256, 256, 0, stream>>>(encoded, enc16, BATCH * HDIM);
    xtab_kernel<<<(VOCAB * G4H + 255) / 256, 256, 0, stream>>>(emb, W_ih, b_ih, b_hh, xtab);
    initstate_kernel<<<(BATCH + 255) / 256, 256, 0, stream>>>(last, stopped);

    // h0 / c0: (B/64)*(H/16)*2 = 32768 waves, 8 waves per block
    proj_kernel<<<32768 / 8, 256, 0, stream>>>(enc16, Wc16, Wh16, bc, bh, hA, cbuf);

    for (int s = 0; s < MAXLEN; ++s) {
        const f16* cur = (s & 1) ? hB : hA;
        f16*       nxt = (s & 1) ? hA : hB;
        // 512 m-groups x 4 j-groups = 2048 blocks, 8 waves each
        lstm_step_kernel<<<2048, 256, 0, stream>>>(cur, nxt, cbuf, Whh16, xtab, last);
        // B/32 = 1024 waves -> 256 blocks of 128 threads
        logits_kernel<<<1024 / 4, 128, 0, stream>>>(nxt, Wa16, ba, last, stopped,
                                                    msgb, lpb, entb, s);
    }

    finalize_kernel<<<(BATCH + 255) / 256, 256, 0, stream>>>(msgb, lpb, entb, out);
}
